// StochasticTwoLayerGCN_26809185861707
// MI455X (gfx1250) — matrix operand order repaired and checked by reading, not compiled, not run on hardware
//
#include <hip/hip_runtime.h>

typedef __attribute__((ext_vector_type(16))) __bf16        v16bf;
typedef __attribute__((ext_vector_type(8)))  float         v8f;
typedef __attribute__((ext_vector_type(4)))  float         v4f;
typedef __attribute__((ext_vector_type(8)))  unsigned int  v8u;

#define DFEAT 128
#define WPACK_ELEMS (4 * 8 * 32 * 8)   // kk(4) x tn(8) x lane(32) x vgpr(8) = 8192 dwords

// ---------------- utility kernels ----------------

__global__ void zero_f32(float* __restrict__ p, long long n) {
  long long i = (long long)blockIdx.x * blockDim.x + threadIdx.x;
  if (i < n) p[i] = 0.0f;
}

__global__ void degree_kernel(const int* __restrict__ src, const int* __restrict__ dst,
                              float* __restrict__ deg_src, float* __restrict__ deg_dst, int E) {
  int e = blockIdx.x * blockDim.x + threadIdx.x;
  if (e < E) {
    atomicAdd(&deg_src[src[e]], 1.0f);
    atomicAdd(&deg_dst[dst[e]], 1.0f);
  }
}

__global__ void norm_kernel(float* __restrict__ p, long long n) {
  long long i = (long long)blockIdx.x * blockDim.x + threadIdx.x;
  if (i < n) p[i] = rsqrtf(fmaxf(p[i], 1.0f));
}

// One thread per (edge, 4 features): b128 gather + 4 f32 atomics.
__global__ void scatter_kernel(const float* __restrict__ x, const int* __restrict__ src,
                               const int* __restrict__ dst, const float* __restrict__ nsrc,
                               float* __restrict__ agg, long long total /* E*32 */) {
  long long idx = (long long)blockIdx.x * blockDim.x + threadIdx.x;
  if (idx >= total) return;
  int e  = (int)(idx >> 5);
  int f4 = (int)(idx & 31) << 2;
  int s = src[e];
  int d = dst[e];
  float nr = nsrc[s];
  v4f xv = *(const v4f*)(x + (long long)s * DFEAT + f4);
  float* base = agg + (long long)d * DFEAT + f4;
  atomicAdd(base + 0, xv[0] * nr);
  atomicAdd(base + 1, xv[1] * nr);
  atomicAdd(base + 2, xv[2] * nr);
  atomicAdd(base + 3, xv[3] * nr);
}

// ---------------- W pre-pack into WMMA B-fragment layout ----------------
// B 32x16 bf16 fragment layout (matches A): lane<16 -> col N=lane, K=0..7,16..23 (2/VGPR);
// lane>=16 -> col N=lane-16, K=8..15,24..31. Packed flat as [kk][tn][lane][vgpr] dwords so each
// lane loads its 8-dword fragment with one contiguous 32B access.
__global__ void pack_w(const float* __restrict__ W, unsigned int* __restrict__ hi,
                       unsigned int* __restrict__ lo) {
  int idx = blockIdx.x * blockDim.x + threadIdx.x;
  if (idx >= WPACK_ELEMS) return;
  int v    = idx & 7;
  int lane = (idx >> 3) & 31;
  int tn   = (idx >> 8) & 7;
  int kk   = idx >> 11;
  int half = lane >> 4;
  int l15  = lane & 15;
  int col  = tn * 16 + l15;
  int k    = kk * 32 + (v >> 2) * 16 + half * 8 + 2 * (v & 3);
  float w0 = W[(long long)k * DFEAT + col];
  float w1 = W[(long long)(k + 1) * DFEAT + col];
  __bf16 h0 = (__bf16)w0;
  __bf16 h1 = (__bf16)w1;
  __bf16 e0 = (__bf16)(w0 - (float)h0);
  __bf16 e1 = (__bf16)(w1 - (float)h1);
  unsigned int uh0 = __builtin_bit_cast(unsigned short, h0);
  unsigned int uh1 = __builtin_bit_cast(unsigned short, h1);
  unsigned int ul0 = __builtin_bit_cast(unsigned short, e0);
  unsigned int ul1 = __builtin_bit_cast(unsigned short, e1);
  hi[idx] = uh0 | (uh1 << 16);
  lo[idx] = ul0 | (ul1 << 16);
}

// ---------------- WMMA GEMM ----------------
// out = relu( (A * ndst[row]) @ W + bias ) with split-precision bf16 WMMA:
//   A = Ah + Al, B = Bh + Bl (prepacked);  D += Ah*Bh + Ah*Bl + Al*Bh  (f32 accumulate)
// One wave owns a 16-row block: A fragments built once, reused for all 8 column tiles.

__device__ inline void split_bf16(float x, __bf16& hi, __bf16& lo) {
  __bf16 h = (__bf16)x;
  hi = h;
  lo = (__bf16)(x - (float)h);
}

__global__ void gcn_gemm_wmma(const float* __restrict__ A, const float* __restrict__ ndst,
                              const unsigned int* __restrict__ Whi,
                              const unsigned int* __restrict__ Wlo,
                              const float* __restrict__ bias,
                              float* __restrict__ out, int N) {
  const int lane = threadIdx.x & 31;
  const int wave = threadIdx.x >> 5;           // 4 waves / block
  const int tm   = blockIdx.x * 4 + wave;      // 16-row block index
  const int tiles_m = N / 16;
  if (tm >= tiles_m) return;

  const int half = lane >> 4;
  const int l15  = lane & 15;
  const int rowA = tm * 16 + l15;              // A: lane holds one row
  const float nrm = ndst[rowA];
  const float* Arow = A + (long long)rowA * DFEAT;

  // Build all four K-step A fragments (hi/lo) once.
  v16bf ahi[4], alo[4];
  #pragma unroll
  for (int kk = 0; kk < 4; ++kk) {
    const int kb = kk * 32;
    // A 16x32 bf16 layout: per-lane K set is two contiguous 8-float runs.
    v4f f0 = *(const v4f*)(Arow + kb + half * 8);
    v4f f1 = *(const v4f*)(Arow + kb + half * 8 + 4);
    v4f f2 = *(const v4f*)(Arow + kb + 16 + half * 8);
    v4f f3 = *(const v4f*)(Arow + kb + 16 + half * 8 + 4);
    #pragma unroll
    for (int j = 0; j < 4; ++j) {
      __bf16 h, l;
      split_bf16(f0[j] * nrm, h, l); ahi[kk][j]      = h; alo[kk][j]      = l;
      split_bf16(f1[j] * nrm, h, l); ahi[kk][4 + j]  = h; alo[kk][4 + j]  = l;
      split_bf16(f2[j] * nrm, h, l); ahi[kk][8 + j]  = h; alo[kk][8 + j]  = l;
      split_bf16(f3[j] * nrm, h, l); ahi[kk][12 + j] = h; alo[kk][12 + j] = l;
    }
  }

  // Rolled loop over the 8 column tiles keeps only one acc + one B pair live.
  #pragma unroll 1
  for (int tn = 0; tn < 8; ++tn) {
    v8f acc = {};
    #pragma unroll
    for (int kk = 0; kk < 4; ++kk) {
      const int boff = (((kk * 8 + tn) * 32) + lane) << 3;
      v16bf bhi = __builtin_bit_cast(v16bf, *(const v8u*)(Whi + boff));
      v16bf blo = __builtin_bit_cast(v16bf, *(const v8u*)(Wlo + boff));
      acc = __builtin_amdgcn_wmma_f32_16x16x32_bf16(false, ahi[kk], false, bhi, (short)0, acc, false, false);
      acc = __builtin_amdgcn_wmma_f32_16x16x32_bf16(false, ahi[kk], false, blo, (short)0, acc, false, false);
      acc = __builtin_amdgcn_wmma_f32_16x16x32_bf16(false, alo[kk], false, bhi, (short)0, acc, false, false);
    }
    // C/D 16x16 f32 layout: VGPR v holds M = v + 8*half, N = l15.
    const int col = tn * 16 + l15;
    const float bv = bias[col];
    #pragma unroll
    for (int v = 0; v < 8; ++v) {
      long long row = (long long)tm * 16 + v + half * 8;
      out[row * DFEAT + col] = fmaxf(acc[v] + bv, 0.0f);
    }
  }
}

// ---------------- host launcher ----------------

extern "C" void kernel_launch(void* const* d_in, const int* in_sizes, int n_in,
                              void* d_out, int out_size, void* d_ws, size_t ws_size,
                              hipStream_t stream) {
  const float* x  = (const float*)d_in[0];
  const int*  src = (const int*)d_in[1];
  const int*  dst = (const int*)d_in[2];
  const float* W1 = (const float*)d_in[3];
  const float* b1 = (const float*)d_in[4];
  const float* W2 = (const float*)d_in[5];
  const float* b2 = (const float*)d_in[6];

  const int N = in_sizes[0] / DFEAT;
  const int E = in_sizes[1];

  float* ws   = (float*)d_ws;
  float* nsrc = ws;                                  // N floats (deg_out -> norm_src)
  float* ndst = ws + N;                              // N floats (deg_in  -> norm_dst)
  float* agg  = ws + 2LL * N;                        // N*128 floats
  float* h    = agg + (long long)N * DFEAT;          // N*128 floats
  unsigned int* w1hi = (unsigned int*)(h + (long long)N * DFEAT);
  unsigned int* w1lo = w1hi + WPACK_ELEMS;
  unsigned int* w2hi = w1lo + WPACK_ELEMS;
  unsigned int* w2lo = w2hi + WPACK_ELEMS;

  const int tb = 256;
  const long long nd   = (long long)N * DFEAT;
  const long long work = (long long)E * (DFEAT / 4); // 1 thread per (edge, 4 feats)
  const int rowblocks = N / 16;
  const int gemm_blocks = (rowblocks + 3) / 4;       // 4 waves / block

  // degrees -> norms (shared by both layers) ; W pre-pack (independent)
  zero_f32<<<(int)((2LL * N + tb - 1) / tb), tb, 0, stream>>>(nsrc, 2LL * N);
  degree_kernel<<<(E + tb - 1) / tb, tb, 0, stream>>>(src, dst, nsrc, ndst, E);
  norm_kernel<<<(int)((2LL * N + tb - 1) / tb), tb, 0, stream>>>(nsrc, 2LL * N);
  pack_w<<<WPACK_ELEMS / tb, tb, 0, stream>>>(W1, w1hi, w1lo);
  pack_w<<<WPACK_ELEMS / tb, tb, 0, stream>>>(W2, w2hi, w2lo);

  // layer 1
  zero_f32<<<(int)((nd + tb - 1) / tb), tb, 0, stream>>>(agg, nd);
  scatter_kernel<<<(int)((work + tb - 1) / tb), tb, 0, stream>>>(x, src, dst, nsrc, agg, work);
  gcn_gemm_wmma<<<gemm_blocks, 128, 0, stream>>>(agg, ndst, w1hi, w1lo, b1, h, N);

  // layer 2
  zero_f32<<<(int)((nd + tb - 1) / tb), tb, 0, stream>>>(agg, nd);
  scatter_kernel<<<(int)((work + tb - 1) / tb), tb, 0, stream>>>(h, src, dst, nsrc, agg, work);
  gcn_gemm_wmma<<<gemm_blocks, 128, 0, stream>>>(agg, ndst, w2hi, w2lo, b2, (float*)d_out, N);
}
